// GlobalTokenAttention_82282983456968
// MI455X (gfx1250) — compile-verified, matrix operand order checked by reading, not run on hardware
//
#include <hip/hip_runtime.h>
#include <cmath>

// ---------------------------------------------------------------------------
// GlobalTokenAttention forward for MI455X (gfx1250, wave32, WMMA).
// K>=32 contractions -> V_WMMA_F32_16X16X32_F16 with f32 accumulate.
// - LDS tiles stored in WMMA fragment order (v16h operand = 2x ds_load_b128)
// - global staging: float4 loads, software-pipelined against the WMMAs
// - 128x64 block tile, 8 waves, each wave computes 32x32 (4 WMMA / chunk)
// Shapes: B=8, C=128, H=W=64, L=4096, d=256, K=4, N_STATE=16, R=8.
// ---------------------------------------------------------------------------

typedef __attribute__((ext_vector_type(16))) _Float16 v16h;
typedef __attribute__((ext_vector_type(8)))  _Float16 v8h;
typedef __attribute__((ext_vector_type(8)))  float    v8f;

#define WMMA16(Af, Bf, Cacc) \
  __builtin_amdgcn_wmma_f32_16x16x32_f16(false, (Af), false, (Bf), (short)0, (Cacc), false, false)

#define FR 24  // fragment row stride in halves (48B: 16B-aligned, bank-spread)

__device__ __forceinline__ float gta_silu(float x)     { return x / (1.f + __expf(-x)); }
__device__ __forceinline__ float gta_softplus(float x) { return (x > 20.f) ? x : __logf(1.f + __expf(x)); }

__device__ __forceinline__ v8h pack8(float4 a, float4 b) {
  v8h h;
  h[0] = (_Float16)a.x; h[1] = (_Float16)a.y; h[2] = (_Float16)a.z; h[3] = (_Float16)a.w;
  h[4] = (_Float16)b.x; h[5] = (_Float16)b.y; h[6] = (_Float16)b.z; h[7] = (_Float16)b.w;
  return h;
}

__device__ __forceinline__ v16h load_frag(const _Float16* p) {
  union { uint4 q[2]; v16h h; } u;
  u.q[0] = *(const uint4*)(p);
  u.q[1] = *(const uint4*)(p + 8);
  return u.h;
}

// Fragment-order indexing (ISA 7.12.2, 16-bit 16x32 operand):
//   lane = (m|n) + 16*hk,  hk = (kk>>3)&1,  element e = (kk&7) | ((kk&16)>>1)

// ---------------------------------------------------------------------------
// Generic strided, batched GEMM:  C[m,n] = sum_k A[m,k]*B[k,n] (+bias[n])
// Block tile 128x64, software-pipelined staging.
// AKC: A unit-stride along K. BNC: B unit-stride along N. NGUARD: N%64 != 0.
// Requires M%128==0 and Kd%32==0 (true for every call site here).
// ---------------------------------------------------------------------------
template<bool AKC, bool BNC, bool NGUARD>
__global__ __launch_bounds__(256)
void gta_gemm_wmma(const float* __restrict__ A, long Abat, long Asm, long Ask,
                   const float* __restrict__ B, long Bbat, long Bsk, long Bsn,
                   float* __restrict__ C, long Cbat, long Csm, long Csn,
                   int M, int N, int Kd, const float* __restrict__ bias)
{
  __shared__ _Float16 Af[8][32][FR];
  __shared__ _Float16 Bf[4][32][FR];
  A += (long)blockIdx.z * Abat;
  B += (long)blockIdx.z * Bbat;
  C += (long)blockIdx.z * Cbat;
  const int m0 = blockIdx.y * 128;
  const int n0 = blockIdx.x * 64;
  const int tid = threadIdx.x;
  const int lane = tid & 31;
  const int wv = tid >> 5, wr = wv >> 1, wc = wv & 1;

  v8f acc00 = {}, acc01 = {}, acc10 = {}, acc11 = {};

  // register staging buffers (prefetch target)
  v8h   aH[2]; float aF[2][8];
  v8h   bH;    float bF[8];

  auto fetchA = [&](int k0) {
    if constexpr (AKC) {
#pragma unroll
      for (int it = 0; it < 2; ++it) {
        int task = tid + it * 256;             // 128 rows x 4 kgroups
        int r = task >> 2, g = task & 3;
        const float* src = A + (long)(m0 + r) * Asm + (k0 + g * 8);
        float4 p0 = *(const float4*)src;
        float4 p1 = *(const float4*)(src + 4);
        aH[it] = pack8(p0, p1);
      }
    } else {                                   // A unit-stride along M
#pragma unroll
      for (int it = 0; it < 2; ++it) {
        int task = tid + it * 256;             // 32 kk x 16 mgroups
        int kk = task >> 4, mg = task & 15;
        const float* src = A + (long)(k0 + kk) * Ask + (m0 + mg * 8);
        float4 p0 = *(const float4*)src;
        float4 p1 = *(const float4*)(src + 4);
        aF[it][0] = p0.x; aF[it][1] = p0.y; aF[it][2] = p0.z; aF[it][3] = p0.w;
        aF[it][4] = p1.x; aF[it][5] = p1.y; aF[it][6] = p1.z; aF[it][7] = p1.w;
      }
    }
  };
  auto storeA = [&]() {
    if constexpr (AKC) {
#pragma unroll
      for (int it = 0; it < 2; ++it) {
        int task = tid + it * 256;
        int r = task >> 2, g = task & 3;
        *(v8h*)&Af[r >> 4][(r & 15) + ((g & 1) << 4)][(g >> 1) << 3] = aH[it];
      }
    } else {
#pragma unroll
      for (int it = 0; it < 2; ++it) {
        int task = tid + it * 256;
        int kk = task >> 4, mg = task & 15;
        const int hk = (kk >> 3) & 1, e = (kk & 7) | ((kk & 16) >> 1);
#pragma unroll
        for (int i = 0; i < 8; ++i) {
          int m = mg * 8 + i;
          Af[m >> 4][(m & 15) + (hk << 4)][e] = (_Float16)aF[it][i];
        }
      }
    }
  };
  auto fetchB = [&](int k0) {
    if constexpr (BNC) {
      const int kk = tid >> 3, ng = tid & 7;
      const int gn = n0 + ng * 8;
      float4 p0 = {}, p1 = {};
      if (!NGUARD || gn < N) {
        const float* src = B + (long)(k0 + kk) * Bsk + gn;
        p0 = *(const float4*)src; p1 = *(const float4*)(src + 4);
      }
      bF[0] = p0.x; bF[1] = p0.y; bF[2] = p0.z; bF[3] = p0.w;
      bF[4] = p1.x; bF[5] = p1.y; bF[6] = p1.z; bF[7] = p1.w;
    } else {                                   // B unit-stride along K
      const int cn = tid >> 2, kg = tid & 3;
      float4 p0 = {}, p1 = {};
      if (!NGUARD || (n0 + cn) < N) {
        const float* src = B + (long)(n0 + cn) * Bsn + (k0 + kg * 8);
        p0 = *(const float4*)src; p1 = *(const float4*)(src + 4);
      }
      bH = pack8(p0, p1);
    }
  };
  auto storeB = [&]() {
    if constexpr (BNC) {
      const int kk = tid >> 3, ng = tid & 7;
      const int hk = (kk >> 3) & 1, e = (kk & 7) | ((kk & 16) >> 1);
#pragma unroll
      for (int i = 0; i < 8; ++i) {
        int cn = ng * 8 + i;
        Bf[cn >> 4][(cn & 15) + (hk << 4)][e] = (_Float16)bF[i];
      }
    } else {
      const int cn = tid >> 2, kg = tid & 3;
      *(v8h*)&Bf[cn >> 4][(cn & 15) + ((kg & 1) << 4)][(kg >> 1) << 3] = bH;
    }
  };

  fetchA(0); fetchB(0);
  for (int k0 = 0; k0 < Kd; k0 += 32) {
    storeA(); storeB();
    __syncthreads();
    if (k0 + 32 < Kd) { fetchA(k0 + 32); fetchB(k0 + 32); }  // overlap with WMMA
    v16h af0 = load_frag(&Af[wr * 2][lane][0]);
    v16h af1 = load_frag(&Af[wr * 2 + 1][lane][0]);
    v16h bf0 = load_frag(&Bf[wc * 2][lane][0]);
    v16h bf1 = load_frag(&Bf[wc * 2 + 1][lane][0]);
    acc00 = WMMA16(af0, bf0, acc00);
    acc01 = WMMA16(af0, bf1, acc01);
    acc10 = WMMA16(af1, bf0, acc10);
    acc11 = WMMA16(af1, bf1, acc11);
    __syncthreads();
  }

  const int cl = lane & 15, rh = lane >> 4;
  const int gn0 = n0 + wc * 32 + cl;
  const int gn1 = gn0 + 16;
  const float b0 = (!NGUARD || gn0 < N) ? (bias ? bias[gn0] : 0.f) : 0.f;
  const float b1 = (!NGUARD || gn1 < N) ? (bias ? bias[gn1] : 0.f) : 0.f;
#pragma unroll
  for (int r = 0; r < 8; ++r) {
    int gmA = m0 + (wr * 2) * 16 + r + 8 * rh;
    int gmB = m0 + (wr * 2 + 1) * 16 + r + 8 * rh;
    if (!NGUARD || gn0 < N) {
      C[(long)gmA * Csm + (long)gn0 * Csn] = acc00[r] + b0;
      C[(long)gmB * Csm + (long)gn0 * Csn] = acc10[r] + b0;
    }
    if (!NGUARD || gn1 < N) {
      C[(long)gmA * Csm + (long)gn1 * Csn] = acc01[r] + b1;
      C[(long)gmB * Csm + (long)gn1 * Csn] = acc11[r] + b1;
    }
  }
}

// ---------------------------------------------------------------------------
// Pre-transpose lepe weights (C,C,3,3) -> wT[k=tap*128+ci][co] (k-major).
// ---------------------------------------------------------------------------
__global__ __launch_bounds__(256)
void gta_transpose_w(const float* __restrict__ w, float* __restrict__ wT)
{
  int idx = blockIdx.x * 256 + threadIdx.x;      // 1152*128
  if (idx >= 1152 * 128) return;
  int co = idx & 127, kk = idx >> 7;
  int ci = kk & 127, tap = kk >> 7;
  wT[(long)kk * 128 + co] = w[((long)co * 128 + ci) * 9 + tap];
}

// ---------------------------------------------------------------------------
// 3x3 conv (lepe), implicit GEMM, 128x64 tile, pipelined, K = 9*128 = 1152
// (36 chunks; chunk c: tap = c>>2 via lookup, ci base = (c&3)*32).
// ---------------------------------------------------------------------------
__global__ __launch_bounds__(256)
void gta_conv3x3_wmma(const float* __restrict__ qkv, const float* __restrict__ wT,
                      const float* __restrict__ bias, float* __restrict__ out)
{
  __shared__ _Float16 Af[8][32][FR];
  __shared__ _Float16 Bf[4][32][FR];
  const int b  = blockIdx.z;
  const int m0 = blockIdx.y * 128;      // pixel l
  const int n0 = blockIdx.x * 64;       // out channel
  const int tid = threadIdx.x, lane = tid & 31, wv = tid >> 5;
  const int wr = wv >> 1, wc = wv & 1;

  static const int KY[9] = {-1,-1,-1, 0,0,0, 1,1,1};
  static const int KX[9] = {-1, 0, 1,-1,0,1,-1,0,1};

  v8f acc00 = {}, acc01 = {}, acc10 = {}, acc11 = {};
  v8h aH[2]; float bF[8];

  auto fetchA = [&](int c) {
    const int tap = c >> 2, cib = (c & 3) * 32;
    const int ky = KY[tap], kx = KX[tap];
#pragma unroll
    for (int it = 0; it < 2; ++it) {
      int task = tid + it * 256;
      int r = task >> 2, g = task & 3;
      int l = m0 + r;
      int yy = (l >> 6) + ky, xx = (l & 63) + kx;
      float4 p0 = {}, p1 = {};
      if (yy >= 0 && yy < 64 && xx >= 0 && xx < 64) {
        const float* src = qkv + ((long)(b * 4096 + yy * 64 + xx)) * 384 + 256 + cib + g * 8;
        p0 = *(const float4*)src; p1 = *(const float4*)(src + 4);
      }
      aH[it] = pack8(p0, p1);
    }
  };
  auto storeA = [&]() {
#pragma unroll
    for (int it = 0; it < 2; ++it) {
      int task = tid + it * 256;
      int r = task >> 2, g = task & 3;
      *(v8h*)&Af[r >> 4][(r & 15) + ((g & 1) << 4)][(g >> 1) << 3] = aH[it];
    }
  };
  auto fetchB = [&](int c) {
    const int k0 = c * 32;
    const int kk = tid >> 3, ng = tid & 7;
    const float* src = wT + (long)(k0 + kk) * 128 + n0 + ng * 8;
    float4 p0 = *(const float4*)src;
    float4 p1 = *(const float4*)(src + 4);
    bF[0] = p0.x; bF[1] = p0.y; bF[2] = p0.z; bF[3] = p0.w;
    bF[4] = p1.x; bF[5] = p1.y; bF[6] = p1.z; bF[7] = p1.w;
  };
  auto storeB = [&]() {
    const int kk = tid >> 3, ng = tid & 7;
    const int hk = (kk >> 3) & 1, e = (kk & 7) | ((kk & 16) >> 1);
#pragma unroll
    for (int i = 0; i < 8; ++i) {
      int cn = ng * 8 + i;
      Bf[cn >> 4][(cn & 15) + (hk << 4)][e] = (_Float16)bF[i];
    }
  };

  fetchA(0); fetchB(0);
  for (int c = 0; c < 36; ++c) {
    storeA(); storeB();
    __syncthreads();
    if (c + 1 < 36) { fetchA(c + 1); fetchB(c + 1); }
    v16h af0 = load_frag(&Af[wr * 2][lane][0]);
    v16h af1 = load_frag(&Af[wr * 2 + 1][lane][0]);
    v16h bf0 = load_frag(&Bf[wc * 2][lane][0]);
    v16h bf1 = load_frag(&Bf[wc * 2 + 1][lane][0]);
    acc00 = WMMA16(af0, bf0, acc00);
    acc01 = WMMA16(af0, bf1, acc01);
    acc10 = WMMA16(af1, bf0, acc10);
    acc11 = WMMA16(af1, bf1, acc11);
    __syncthreads();
  }

  const int cl = lane & 15, rh = lane >> 4;
  const int gn0 = n0 + wc * 32 + cl, gn1 = gn0 + 16;
  const float b0 = bias[gn0], b1 = bias[gn1];
#pragma unroll
  for (int r = 0; r < 8; ++r) {
    int gmA = m0 + (wr * 2) * 16 + r + 8 * rh;
    int gmB = m0 + (wr * 2 + 1) * 16 + r + 8 * rh;
    out[((long)(b * 4096 + gmA)) * 128 + gn0] = acc00[r] + b0;
    out[((long)(b * 4096 + gmA)) * 128 + gn1] = acc01[r] + b1;
    out[((long)(b * 4096 + gmB)) * 128 + gn0] = acc10[r] + b0;
    out[((long)(b * 4096 + gmB)) * 128 + gn1] = acc11[r] + b1;
  }
}

// ---------------------------------------------------------------------------
// Fused 64x64 attention per (b, row h, head n): S=q k^T (WMMA), softmax, P v.
// ---------------------------------------------------------------------------
__global__ __launch_bounds__(256)
void gta_attn_wmma(const float* __restrict__ qkv, float* __restrict__ outY, float scale)
{
  __shared__ _Float16 Qf[4][32][FR];        // A operand of S (rows = w)
  __shared__ _Float16 Kf[4][32][FR];        // B operand of S (cols = w')
  __shared__ _Float16 Vf[2][2][32][FR];     // B operand of Y [kchunk][coltile]
  __shared__ _Float16 Pf[4][2][32][FR];     // A operand of Y [rowtile][kchunk]
  __shared__ float    Sf[64][64];
  const int nh = blockIdx.x, hrow = blockIdx.y, b = blockIdx.z;
  const int tid = threadIdx.x, lane = tid & 31, wv = tid >> 5;

  { // stage q/k/v: thread (r, g): row r, 8 contiguous channels
    const int r = tid >> 2, g = tid & 3;
    const float* src = qkv + ((long)(b * 4096 + hrow * 64 + r)) * 384 + nh * 32 + g * 8;
    float4 q0 = *(const float4*)(src);       float4 q1 = *(const float4*)(src + 4);
    float4 k0 = *(const float4*)(src + 128); float4 k1 = *(const float4*)(src + 132);
    float4 v0 = *(const float4*)(src + 256); float4 v1 = *(const float4*)(src + 260);
    const int fl = (r & 15) + ((g & 1) << 4), eb = (g >> 1) << 3, tl = r >> 4;
    *(v8h*)&Qf[tl][fl][eb] = pack8(q0, q1);
    *(v8h*)&Kf[tl][fl][eb] = pack8(k0, k1);
    float vv[8] = {v0.x, v0.y, v0.z, v0.w, v1.x, v1.y, v1.z, v1.w};
    const int kc = r >> 5, kk = r & 31;
    const int hk = (kk >> 3) & 1, e = (kk & 7) | ((kk & 16) >> 1);
#pragma unroll
    for (int i = 0; i < 8; ++i) {
      int cc = g * 8 + i;
      Vf[kc][cc >> 4][(cc & 15) + (hk << 4)][e] = (_Float16)vv[i];
    }
  }
  __syncthreads();

  { // S = q k^T  (M=N=64, K=32)
    const int wr = wv >> 1, wc = wv & 1;
    v16h af  = load_frag(&Qf[wr][lane][0]);
    v16h bf0 = load_frag(&Kf[wc * 2][lane][0]);
    v16h bf1 = load_frag(&Kf[wc * 2 + 1][lane][0]);
    v8f s0 = {}, s1 = {};
    s0 = WMMA16(af, bf0, s0);
    s1 = WMMA16(af, bf1, s1);
    const int cl = lane & 15, rh = lane >> 4;
#pragma unroll
    for (int r = 0; r < 8; ++r) {
      Sf[wr * 16 + r + 8 * rh][wc * 32 + cl]      = s0[r] * scale;
      Sf[wr * 16 + r + 8 * rh][wc * 32 + 16 + cl] = s1[r] * scale;
    }
  }
  __syncthreads();

  if (tid < 64) {                               // row softmax -> Pf (frag order)
    float mx = -1e30f;
#pragma unroll
    for (int c = 0; c < 64; ++c) mx = fmaxf(mx, Sf[tid][c]);
    float sum = 0.f;
#pragma unroll
    for (int c = 0; c < 64; ++c) { float e = __expf(Sf[tid][c] - mx); sum += e; Sf[tid][c] = e; }
    float inv = 1.f / sum;
    const int tl = tid >> 4, ml = tid & 15;
#pragma unroll
    for (int c = 0; c < 64; ++c) {
      int kk = c & 31;
      int hk = (kk >> 3) & 1, e = (kk & 7) | ((kk & 16) >> 1);
      Pf[tl][c >> 5][ml + (hk << 4)][e] = (_Float16)(Sf[tid][c] * inv);
    }
  }
  __syncthreads();

  { // Y = P v  (M=64, N=32, K=64): 8 tiles, one per wave
    const int wr = wv >> 1, wc = wv & 1;
    v8f y = {};
#pragma unroll
    for (int kc = 0; kc < 2; ++kc) {
      v16h af = load_frag(&Pf[wr][kc][lane][0]);
      v16h bf = load_frag(&Vf[kc][wc][lane][0]);
      y = WMMA16(af, bf, y);
    }
    const int cl = lane & 15, rh = lane >> 4;
#pragma unroll
    for (int r = 0; r < 8; ++r) {
      int wq = wr * 16 + r + 8 * rh;
      int cc = nh * 32 + wc * 16 + cl;
      outY[((long)(b * 4096 + hrow * 64 + wq)) * 128 + cc] = y[r];
    }
  }
}

// ---------------------------------------------------------------------------
// LayerNorm over last dim = 128 (one block per row).
// ---------------------------------------------------------------------------
__global__ __launch_bounds__(128)
void gta_ln128(const float* __restrict__ in, float* __restrict__ out,
               const float* __restrict__ g, const float* __restrict__ b)
{
  __shared__ float red[128];
  const long row = blockIdx.x;
  const int t = threadIdx.x;
  float v = in[row * 128 + t];
  red[t] = v; __syncthreads();
  for (int s = 64; s > 0; s >>= 1) { if (t < s) red[t] += red[t + s]; __syncthreads(); }
  float mu = red[0] * (1.f / 128.f); __syncthreads();
  float dv = v - mu;
  red[t] = dv * dv; __syncthreads();
  for (int s = 64; s > 0; s >>= 1) { if (t < s) red[t] += red[t + s]; __syncthreads(); }
  float var = red[0] * (1.f / 128.f);
  out[row * 128 + t] = dv * rsqrtf(var + 1e-5f) * g[t] + b[t];
}

// ---------------------------------------------------------------------------
// Depthwise 3x3 conv + SiLU. In: xz (B,L,512) chans 0..255. Out: xx (B,d,L).
// ---------------------------------------------------------------------------
__global__ __launch_bounds__(256)
void gta_dwconv_silu(const float* __restrict__ xz, const float* __restrict__ cw,
                     const float* __restrict__ cb, float* __restrict__ xx)
{
  long idx = (long)blockIdx.x * 256 + threadIdx.x;     // B*d*L
  int l = idx & 4095; long t = idx >> 12;
  int dd = (int)(t & 255); int b = (int)(t >> 8);
  int i = l >> 6, j = l & 63;
  float s = cb[dd];
#pragma unroll
  for (int ky = 0; ky < 3; ++ky)
#pragma unroll
    for (int kx = 0; kx < 3; ++kx) {
      int yy = i + ky - 1, xc = j + kx - 1;
      if (yy >= 0 && yy < 64 && xc >= 0 && xc < 64)
        s += xz[((long)(b * 4096 + yy * 64 + xc)) * 512 + dd] * cw[dd * 9 + ky * 3 + kx];
    }
  xx[idx] = gta_silu(s);
}

// ---------------------------------------------------------------------------
// Build xs (B,K,d,L): k0=xx, k1=transpose(H,W), k2=flip(k0), k3=flip(k1).
// ---------------------------------------------------------------------------
__global__ __launch_bounds__(256)
void gta_build_xs(const float* __restrict__ xx, float* __restrict__ xs)
{
  long idx = (long)blockIdx.x * 256 + threadIdx.x;     // B*d*L
  int l = idx & 4095; long t = idx >> 12;
  int dd = (int)(t & 255); int b = (int)(t >> 8);
  long ch = ((long)b * 256 + dd) * 4096;
  int lt  = (l & 63) * 64 + (l >> 6);
  int lr  = 4095 - l;
  int ltr = (lr & 63) * 64 + (lr >> 6);
  long o = (((long)b * 4) * 256 + dd) * 4096 + l;
  long stride = (long)256 * 4096;
  xs[o]              = xx[ch + l];
  xs[o + stride]     = xx[ch + lt];
  xs[o + 2 * stride] = xx[ch + lr];
  xs[o + 3 * stride] = xx[ch + ltr];
}

// ---------------------------------------------------------------------------
// delta = softplus(dt_w @ dts + dt_b). xdbl (B,K,L,40) rows 0..7 = dts.
// ---------------------------------------------------------------------------
__global__ __launch_bounds__(256)
void gta_dt_proj(const float* __restrict__ xdbl, const float* __restrict__ dtw,
                 const float* __restrict__ dtb, float* __restrict__ delta)
{
  long idx = (long)blockIdx.x * 256 + threadIdx.x;     // B*K*d*L
  int l = idx & 4095; long t = idx >> 12;
  int dd = (int)(t & 255); int bk = (int)(t >> 8);
  int k = bk & 3;
  const float* xr = xdbl + ((long)bk * 4096 + l) * 40;
  const float* wr = dtw + ((long)(k * 256 + dd)) * 8;
  float s = dtb[k * 256 + dd];
#pragma unroll
  for (int r = 0; r < 8; ++r) s += wr[r] * xr[r];
  delta[idx] = gta_softplus(s);
}

// ---------------------------------------------------------------------------
// Selective scan: one lane per (b,k,d) channel, h[16] in VGPRs.
// ---------------------------------------------------------------------------
__global__ __launch_bounds__(256)
void gta_scan(const float* __restrict__ delta, const float* __restrict__ xs,
              const float* __restrict__ xdbl, const float* __restrict__ A_logs,
              float* __restrict__ ys)
{
  const int bk = blockIdx.x;          // b*4 + k
  const int dd = threadIdx.x;         // 0..255
  const int k  = bk & 3;
  float Arow[16], h[16];
#pragma unroll
  for (int n = 0; n < 16; ++n) {
    Arow[n] = -__expf(A_logs[((long)(k * 256 + dd)) * 16 + n]);
    h[n] = 0.f;
  }
  const long ch = ((long)bk * 256 + dd) * 4096;
  const long lb = (long)bk * 4096;
  for (int l = 0; l < 4096; ++l) {
    float dt = delta[ch + l];
    float dx = dt * xs[ch + l];
    const float* bc = xdbl + (lb + l) * 40;
    float4 bq[4], cq[4];
#pragma unroll
    for (int q = 0; q < 4; ++q) {
      bq[q] = ((const float4*)(bc + 8))[q];
      cq[q] = ((const float4*)(bc + 24))[q];
    }
    const float* Bt = (const float*)bq;
    const float* Ct = (const float*)cq;
    float y = 0.f;
#pragma unroll
    for (int n = 0; n < 16; ++n) {
      h[n] = h[n] * __expf(dt * Arow[n]) + dx * Bt[n];
      y += h[n] * Ct[n];
    }
    ys[ch + l] = y;
  }
}

// ---------------------------------------------------------------------------
// Merge 4 scan directions (+D*xs skip), LayerNorm(256), gate with silu(z).
// ---------------------------------------------------------------------------
__global__ __launch_bounds__(256)
void gta_merge_ln_gate(const float* __restrict__ ys, const float* __restrict__ xs,
                       const float* __restrict__ Ds, const float* __restrict__ g,
                       const float* __restrict__ bb, const float* __restrict__ xz,
                       float* __restrict__ ypre)
{
  __shared__ float red[256];
  const int d = threadIdx.x;
  const long bl = blockIdx.x;
  const int b = (int)(bl >> 12), l = (int)(bl & 4095);
  const int lt = (l & 63) * 64 + (l >> 6);

  auto term = [&](int k, int ll) {
    long o = (((long)b * 4 + k) * 256 + d) * 4096 + ll;
    return ys[o] + Ds[k * 256 + d] * xs[o];
  };
  float v = term(0, l) + term(2, 4095 - l) + term(1, lt) + term(3, 4095 - lt);

  red[d] = v; __syncthreads();
  for (int s = 128; s > 0; s >>= 1) { if (d < s) red[d] += red[d + s]; __syncthreads(); }
  float mu = red[0] * (1.f / 256.f); __syncthreads();
  float dv = v - mu;
  red[d] = dv * dv; __syncthreads();
  for (int s = 128; s > 0; s >>= 1) { if (d < s) red[d] += red[d + s]; __syncthreads(); }
  float var = red[0] * (1.f / 256.f);
  float lnv = dv * rsqrtf(var + 1e-5f) * g[d] + bb[d];
  float z = xz[bl * 512 + 256 + d];
  ypre[bl * 256 + d] = lnv * gta_silu(z);
}

// ---------------------------------------------------------------------------
// ytot = attnY + lepe + ss2d_out
// ---------------------------------------------------------------------------
__global__ __launch_bounds__(256)
void gta_add3(const float* __restrict__ a, const float* __restrict__ b,
              const float* __restrict__ c, float* __restrict__ o, long n)
{
  long i = (long)blockIdx.x * 256 + threadIdx.x;
  if (i < n) o[i] = a[i] + b[i] + c[i];
}

// ---------------------------------------------------------------------------

extern "C" void kernel_launch(void* const* d_in, const int* in_sizes, int n_in,
                              void* d_out, int out_size, void* d_ws, size_t ws_size,
                              hipStream_t stream)
{
  (void)in_sizes; (void)n_in; (void)out_size; (void)ws_size;
  const float* x         = (const float*)d_in[0];
  const float* qkv_w     = (const float*)d_in[1];
  const float* qkv_b     = (const float*)d_in[2];
  const float* proj_w    = (const float*)d_in[3];
  const float* proj_b    = (const float*)d_in[4];
  const float* lepe_w    = (const float*)d_in[5];
  const float* lepe_b    = (const float*)d_in[6];
  const float* ln_g      = (const float*)d_in[7];
  const float* ln_b      = (const float*)d_in[8];
  const float* in_proj_w = (const float*)d_in[9];
  const float* conv_w    = (const float*)d_in[10];
  const float* conv_b    = (const float*)d_in[11];
  const float* x_proj_w  = (const float*)d_in[12];
  const float* dt_w      = (const float*)d_in[13];
  const float* dt_b      = (const float*)d_in[14];
  const float* A_logs    = (const float*)d_in[15];
  const float* Ds        = (const float*)d_in[16];
  const float* onorm_g   = (const float*)d_in[17];
  const float* onorm_b   = (const float*)d_in[18];
  const float* out_proj_w= (const float*)d_in[19];
  float* out = (float*)d_out;

  const long B = 8, C = 128, L = 4096, D = 256, K4 = 4;

  float* p = (float*)d_ws;
  float* qkv   = p; p += B * L * 384;      // (B,L,384)
  float* attY  = p; p += B * L * C;        // (B,L,C)
  float* lepe  = p; p += B * L * C;        // (B,L,C)
  float* lnl   = p; p += B * L * C;        // (B,L,C)
  float* xz    = p; p += B * L * 512;      // (B,L,512)
  float* xx    = p; p += B * D * L;        // (B,d,L)
  float* xs    = p; p += B * K4 * D * L;   // (B,K,d,L)
  float* xdbl  = p; p += B * K4 * L * 40;  // (B,K,L,40)
  float* delta = p; p += B * K4 * D * L;   // (B,K,d,L)
  float* ysb   = p; p += B * K4 * D * L;   // (B,K,d,L)
  float* ypre  = p; p += B * L * D;        // (B,L,d)
  float* ssout = p; p += B * L * C;        // (B,L,C)
  float* ytot  = p; p += B * L * C;        // (B,L,C)
  float* wT    = p; p += 1152 * 128;       // transposed lepe weights

  const dim3 blk(256);

  // 0) lepe weight transpose (k-major for contiguous B staging)
  gta_transpose_w<<<dim3(576), blk, 0, stream>>>(lepe_w, wT);

  // 1) qkv 1x1 conv: per b, [L,128] x [128,384] + bias.  A: m contig, B: k contig
  gta_gemm_wmma<false, false, false><<<dim3(6, 32, 8), blk, 0, stream>>>(
      x, C * L, 1, L,  qkv_w, 0, 1, 128,  qkv, L * 384, 384, 1,
      4096, 384, 128, qkv_b);

  // 2) windowed attention per (b, row, head)
  gta_attn_wmma<<<dim3(4, 64, 8), blk, 0, stream>>>(qkv, attY, 1.0f / sqrtf(128.0f));

  // 3) lepe 3x3 conv (implicit GEMM, K=1152)
  gta_conv3x3_wmma<<<dim3(2, 32, 8), blk, 0, stream>>>(qkv, wT, lepe_b, lepe);

  // 4) layernorm(C=128) on lepe -> ss2d input
  gta_ln128<<<dim3(32768), dim3(128), 0, stream>>>(lepe, lnl, ln_g, ln_b);

  // 5) in_proj: [32768,128] x [128,512].  A: k contig, B: n contig
  gta_gemm_wmma<true, true, false><<<dim3(8, 256, 1), blk, 0, stream>>>(
      lnl, 0, 128, 1,  in_proj_w, 0, 512, 1,  xz, 0, 512, 1,
      32768, 512, 128, nullptr);

  // 6) depthwise 3x3 + silu -> xx (B,d,L)
  gta_dwconv_silu<<<dim3(32768), blk, 0, stream>>>(xz, conv_w, conv_b, xx);

  // 7) build 4 scan directions
  gta_build_xs<<<dim3(32768), blk, 0, stream>>>(xx, xs);

  // 8) x_proj per direction k: per b, [L,256] x [256,40].  A: m contig, B: k contig
  for (int k = 0; k < 4; ++k) {
    gta_gemm_wmma<false, false, true><<<dim3(1, 32, 8), blk, 0, stream>>>(
        xs + (long)k * D * L, K4 * D * L, 1, L,
        x_proj_w + (long)k * 40 * D, 0, 1, D,
        xdbl + (long)k * L * 40, K4 * L * 40, 40, 1,
        4096, 40, 256, nullptr);
  }

  // 9) dt projection + softplus -> delta (B,K,d,L)
  gta_dt_proj<<<dim3(131072), blk, 0, stream>>>(xdbl, dt_w, dt_b, delta);

  // 10) selective scan: 32 blocks x 256 lanes = 8192 channels
  gta_scan<<<dim3(32), blk, 0, stream>>>(delta, xs, xdbl, A_logs, ysb);

  // 11) merge directions + out_norm LN(256) + silu(z) gate -> ypre (B,L,d)
  gta_merge_ln_gate<<<dim3(32768), blk, 0, stream>>>(ysb, xs, Ds, onorm_g, onorm_b, xz, ypre);

  // 12) out_proj: [32768,256] x [256,128].  A: k contig, B: n contig
  gta_gemm_wmma<true, true, false><<<dim3(2, 256, 1), blk, 0, stream>>>(
      ypre, 0, 256, 1,  out_proj_w, 0, 128, 1,  ssout, 0, 128, 1,
      32768, 128, 256, nullptr);

  // 13) y = attnY + lepe + ss2d_out
  gta_add3<<<dim3(16384), blk, 0, stream>>>(attY, lepe, ssout, ytot, B * L * C);

  // 14) final 1x1 proj, strided store -> (B,C,H,W).  A: k contig, B: k contig
  gta_gemm_wmma<true, false, false><<<dim3(2, 32, 8), blk, 0, stream>>>(
      ytot, L * C, 128, 1,  proj_w, 0, 1, 128,  out, C * L, 1, L,
      4096, 128, 128, proj_b);
}
